// LstmDecoderWithAttention_10857677324747
// MI455X (gfx1250) — matrix operand order repaired and checked by reading, not compiled, not run on hardware
//
#include <hip/hip_runtime.h>
#include <hip/hip_bf16.h>

typedef __attribute__((ext_vector_type(16))) _Float16 v16h;
typedef __attribute__((ext_vector_type(8)))  _Float16 v8h;
typedef __attribute__((ext_vector_type(8)))  float    v8f;

#define B_  128
#define S_  128
#define T_  32
#define E_  256
#define HS_ 512
#define HT_ 512
#define V_  128
#define G4_ 2048   // 4*HT

// ---------------------------------------------------------------------------
// Generic WMMA GEMM: C[M,N] = A[M,K](f16,row-major) * Bp(packed f16 fragments)
// Bp layout: [ntile][ktile][lane][16 halves]  (ISA 7.12.2 16-bit B layout)
// One wave per 16x16 output tile; tile selection is wave-uniform so EXEC==all-1s.
// ---------------------------------------------------------------------------
template <typename OutT>
__global__ void wmma_gemm_kernel(const _Float16* __restrict__ A,
                                 const _Float16* __restrict__ Bp,
                                 OutT* __restrict__ C,
                                 int M, int N, int K) {
    const int lane  = threadIdx.x & 31;
    const int wib   = threadIdx.x >> 5;
    const int ntiles = N >> 4;
    const int ktiles = K >> 5;
    const int tile = blockIdx.x * 4 + wib;
    const int totalTiles = (M >> 4) * ntiles;
    if (tile >= totalTiles) return;           // uniform per wave
    const int mt = tile / ntiles;
    const int nt = tile - mt * ntiles;
    const int hi = lane >> 4;                 // 0 or 1
    const int lo = lane & 15;
    // A fragment: lane(lo) = row M; hi selects K-halves {hi*8..+7, 16+hi*8..+7}
    const _Float16* arow = A + (size_t)((mt << 4) + lo) * K + hi * 8;
    const _Float16* bptr = Bp + ((size_t)nt * ktiles * 32 + lane) * 16;
    v8f acc = {};
    for (int kt = 0; kt < ktiles; ++kt) {
        v8h alo = *(const v8h*)(arow + kt * 32);
        v8h ahi = *(const v8h*)(arow + kt * 32 + 16);
        v16h a;
#pragma unroll
        for (int i = 0; i < 8; ++i) { a[i] = alo[i]; a[i + 8] = ahi[i]; }
        v16h b = *(const v16h*)(bptr + (size_t)kt * 32 * 16);
        acc = __builtin_amdgcn_wmma_f32_16x16x32_f16(false, a, false, b,
                                                     (short)0, acc, false, false);
    }
    // C/D layout: VGPR v -> row mt*16 + hi*8 + v, col nt*16 + lo
    OutT* crow = C + (size_t)((mt << 4) + (hi << 3)) * N + (nt << 4) + lo;
#pragma unroll
    for (int v = 0; v < 8; ++v) crow[(size_t)v * N] = (OutT)acc[v];
}

// ---------------------------------------------------------------------------
// Pack f32 weights into WMMA B fragments (f16).
// knMajor=0: W is [N, Ksub] row-major (use W[n*Ksub+k]) -> computes X @ W^T
// knMajor=1: W is [Ksub, N] row-major (use W[k*N+n])
// dst fragment index uses ktTotal/ktOff so concatenated-K weights can share one Bp.
// Lane layout: lane(lo)=column n; hi selects K rows {hi*8..+7, 16+hi*8..+7}.
// ---------------------------------------------------------------------------
__global__ void pack_b_kernel(const float* __restrict__ W, _Float16* __restrict__ Bp,
                              int N, int Ksub, int ktTotal, int ktOff, int knMajor) {
    int idx = blockIdx.x * blockDim.x + threadIdx.x;
    int ktSub = Ksub >> 5;
    int total = (N >> 4) * ktSub * 32;
    if (idx >= total) return;
    int lane = idx & 31;
    int t    = idx >> 5;
    int ktL  = t % ktSub;
    int nt   = t / ktSub;
    int n    = (nt << 4) + (lane & 15);
    int kb   = (ktL << 5) + ((lane >> 4) << 3);
    _Float16* out = Bp + ((size_t)(nt * ktTotal + ktOff + ktL) * 32 + lane) * 16;
#pragma unroll
    for (int j = 0; j < 16; ++j) {
        int k = kb + (j < 8 ? j : j + 8);
        float v = knMajor ? W[(size_t)k * N + n] : W[(size_t)n * Ksub + k];
        out[j] = (_Float16)v;
    }
}

__global__ void f32_to_f16_kernel(const float* __restrict__ in,
                                  _Float16* __restrict__ out, int n) {
    int i = blockIdx.x * blockDim.x + threadIdx.x;
    if (i < n) out[i] = (_Float16)in[i];
}

// Copy initial h/c state into workspace (c: f32, h: f16 into GEMM A buffers).
__global__ void init_state_kernel(const float* __restrict__ h0, const float* __restrict__ c0,
                                  float* __restrict__ cst,
                                  _Float16* __restrict__ H0f16, _Float16* __restrict__ Xcat1) {
    int idx = blockIdx.x * blockDim.x + threadIdx.x;
    if (idx >= B_ * HT_) return;
    int b = idx >> 9, h = idx & (HT_ - 1);
    cst[idx]             = c0[idx];             // layer 0
    cst[B_ * HT_ + idx]  = c0[B_ * HT_ + idx];  // layer 1
    H0f16[idx] = (_Float16)h0[idx];             // h[0] -> A of gate GEMM 0
    Xcat1[b * 1024 + 512 + h] = (_Float16)h0[B_ * HT_ + idx]; // h[1] -> A of gate GEMM 1
}

__device__ __forceinline__ float sigmoidf_(float x) { return 1.f / (1.f + __expf(-x)); }

// LSTM cell layer 0: gates = Ghh (GEMM out) + P_vocab[tok] + b_ih0 + b_hh0
__global__ void cell0_kernel(const float* __restrict__ G, const float* __restrict__ Pvoc,
                             const int* __restrict__ target, const int* __restrict__ sot,
                             const float* __restrict__ bih, const float* __restrict__ bhh,
                             float* __restrict__ cst,
                             _Float16* __restrict__ H0f16, _Float16* __restrict__ Xcat1,
                             int t) {
    int idx = blockIdx.x * blockDim.x + threadIdx.x;
    if (idx >= B_ * HT_) return;
    int b = idx >> 9, h = idx & (HT_ - 1);
    int tok = (t == 0) ? sot[0] : target[(t - 1) * B_ + b];
    const float* p = Pvoc + (size_t)tok * G4_;
    const float* g = G + (size_t)b * G4_;
    float gi = g[h]          + p[h]          + bih[h]          + bhh[h];
    float gf = g[512 + h]    + p[512 + h]    + bih[512 + h]    + bhh[512 + h];
    float gg = g[1024 + h]   + p[1024 + h]   + bih[1024 + h]   + bhh[1024 + h];
    float go = g[1536 + h]   + p[1536 + h]   + bih[1536 + h]   + bhh[1536 + h];
    float i = sigmoidf_(gi), f = sigmoidf_(gf), o = sigmoidf_(go);
    float c2 = f * cst[idx] + i * tanhf(gg);
    float h2 = o * tanhf(c2);
    cst[idx] = c2;
    H0f16[idx] = (_Float16)h2;                 // next step's gate-0 A
    Xcat1[b * 1024 + h] = (_Float16)h2;        // this step's gate-1 A (first half)
}

// LSTM cell layer 1
__global__ void cell1_kernel(const float* __restrict__ G,
                             const float* __restrict__ bih, const float* __restrict__ bhh,
                             float* __restrict__ cst,
                             _Float16* __restrict__ Xcat1, _Float16* __restrict__ XcatH,
                             float* __restrict__ htf32) {
    int idx = blockIdx.x * blockDim.x + threadIdx.x;
    if (idx >= B_ * HT_) return;
    int b = idx >> 9, h = idx & (HT_ - 1);
    const float* g = G + (size_t)b * G4_;
    float gi = g[h]        + bih[h]        + bhh[h];
    float gf = g[512 + h]  + bih[512 + h]  + bhh[512 + h];
    float gg = g[1024 + h] + bih[1024 + h] + bhh[1024 + h];
    float go = g[1536 + h] + bih[1536 + h] + bhh[1536 + h];
    float i = sigmoidf_(gi), f = sigmoidf_(gf), o = sigmoidf_(go);
    float c2 = f * cst[B_ * HT_ + idx] + i * tanhf(gg);
    float h2 = o * tanhf(c2);
    cst[B_ * HT_ + idx] = c2;
    Xcat1[b * 1024 + 512 + h] = (_Float16)h2;  // next step's gate-1 A (second half)
    XcatH[b * 1024 + h] = (_Float16)h2;        // hidden GEMM A (ht half)
    htf32[idx] = h2;                           // attention scores
}

// scores[s,b] = sum_t Whs[s,b,t]*ht[b,t]; one wave per (s,b); mask applied.
__global__ void scores_kernel(const _Float16* __restrict__ Whs, const float* __restrict__ ht,
                              const unsigned char* __restrict__ mask, float* __restrict__ sc) {
    int wid = (blockIdx.x * blockDim.x + threadIdx.x) >> 5;
    int lane = threadIdx.x & 31;
    if (wid >= S_ * B_) return;
    int b = wid & (B_ - 1);
    const _Float16* wr = Whs + (size_t)wid * HT_;
    const float* hr = ht + (size_t)b * HT_;
    float acc = 0.f;
#pragma unroll
    for (int i = 0; i < HT_ / 32; ++i) acc += (float)wr[lane + 32 * i] * hr[lane + 32 * i];
#pragma unroll
    for (int off = 16; off; off >>= 1) acc += __shfl_xor(acc, off, 32);
    if (lane == 0) sc[wid] = mask[wid] ? acc : -9999.9f;
}

__global__ void softmax_kernel(const float* __restrict__ sc, float* __restrict__ almt) {
    int b = blockIdx.x, s = threadIdx.x;  // 128 threads = S
    __shared__ float red[S_];
    float v = sc[s * B_ + b];
    red[s] = v; __syncthreads();
    for (int off = 64; off; off >>= 1) { if (s < off) red[s] = fmaxf(red[s], red[s + off]); __syncthreads(); }
    float mx = red[0]; __syncthreads();
    float e = __expf(v - mx);
    red[s] = e; __syncthreads();
    for (int off = 64; off; off >>= 1) { if (s < off) red[s] += red[s + off]; __syncthreads(); }
    almt[s * B_ + b] = e / red[0];
}

// ctx (f16 into XcatH second half) and ctx_emb (f32)
__global__ void ctx_kernel(const float* __restrict__ almt, const float* __restrict__ srcout,
                           const float* __restrict__ srcemb,
                           _Float16* __restrict__ XcatH, float* __restrict__ ctx_emb) {
    int b = blockIdx.x, tid = threadIdx.x;  // 256 threads
    __shared__ float w[S_];
    if (tid < S_) w[tid] = almt[tid * B_ + b];
    __syncthreads();
    float a0 = 0.f, a1 = 0.f, ae = 0.f;
    for (int s = 0; s < S_; ++s) {
        float ws = w[s];
        const float* so = srcout + ((size_t)s * B_ + b) * HS_;
        a0 += ws * so[tid];
        a1 += ws * so[tid + 256];
        ae += ws * srcemb[((size_t)s * B_ + b) * E_ + tid];
    }
    XcatH[b * 1024 + 512 + tid]       = (_Float16)a0;
    XcatH[b * 1024 + 512 + 256 + tid] = (_Float16)a1;
    ctx_emb[b * E_ + tid] = ae;
}

// Norm-controlled residual; emits f16 A for the logits GEMM.
__global__ void residual_kernel(const float* __restrict__ hidraw, const float* __restrict__ bhid,
                                const float* __restrict__ ctx_emb, _Float16* __restrict__ hidres) {
    int b = blockIdx.x, tid = threadIdx.x;  // 256 threads = E
    __shared__ float r1[E_], r2[E_];
    float hc = hidraw[b * E_ + tid] + bhid[tid];
    float ce = ctx_emb[b * E_ + tid];
    r1[tid] = hc * hc; r2[tid] = ce * ce; __syncthreads();
    for (int off = 128; off; off >>= 1) {
        if (tid < off) { r1[tid] += r1[tid + off]; r2[tid] += r2[tid + off]; }
        __syncthreads();
    }
    float na = sqrtf(r1[0]), nb = sqrtf(r2[0]);
    float adj = fminf(na, nb * 0.2f);
    float res = ce + hc / fmaxf(na, 1e-12f) * adj;
    hidres[b * E_ + tid] = (_Float16)res;
}

__global__ void logsoftmax_kernel(const float* __restrict__ logits, float* __restrict__ out, int t) {
    int b = blockIdx.x, v = threadIdx.x;  // 128 threads = V
    __shared__ float red[V_];
    float x = logits[b * V_ + v];
    red[v] = x; __syncthreads();
    for (int off = 64; off; off >>= 1) { if (v < off) red[v] = fmaxf(red[v], red[v + off]); __syncthreads(); }
    float mx = red[0]; __syncthreads();
    float e = __expf(x - mx);
    red[v] = e; __syncthreads();
    for (int off = 64; off; off >>= 1) { if (v < off) red[v] += red[v + off]; __syncthreads(); }
    out[((size_t)t * B_ + b) * V_ + v] = x - mx - logf(red[0]);
}

// ---------------------------------------------------------------------------
template <typename OutT>
static inline void launch_gemm(const _Float16* A, const _Float16* Bp, OutT* C,
                               int M, int N, int K, hipStream_t s) {
    int tiles = (M / 16) * (N / 16);
    int blocks = (tiles + 3) / 4;
    wmma_gemm_kernel<OutT><<<blocks, 128, 0, s>>>(A, Bp, C, M, N, K);
}

extern "C" void kernel_launch(void* const* d_in, const int* in_sizes, int n_in,
                              void* d_out, int out_size, void* d_ws, size_t ws_size,
                              hipStream_t stream) {
    (void)in_sizes; (void)n_in; (void)out_size; (void)ws_size;
    const int*   sot      = (const int*)d_in[0];
    const int*   target   = (const int*)d_in[1];
    const float* src_emb  = (const float*)d_in[2];
    const float* src_out  = (const float*)d_in[3];
    const unsigned char* mask = (const unsigned char*)d_in[4];
    const float* h0       = (const float*)d_in[5];
    const float* c0       = (const float*)d_in[6];
    const float* emb      = (const float*)d_in[7];
    const float* W_ih0    = (const float*)d_in[8];
    const float* W_hh0    = (const float*)d_in[9];
    const float* b_ih0    = (const float*)d_in[10];
    const float* b_hh0    = (const float*)d_in[11];
    const float* W_ih1    = (const float*)d_in[12];
    const float* W_hh1    = (const float*)d_in[13];
    const float* b_ih1    = (const float*)d_in[14];
    const float* b_hh1    = (const float*)d_in[15];
    const float* Wa       = (const float*)d_in[16];
    const float* W_hidden = (const float*)d_in[17];
    const float* b_hidden = (const float*)d_in[18];
    float* out = (float*)d_out;

    // Workspace carve-out (256B aligned)
    char* wsp = (char*)d_ws;
    size_t off = 0;
    auto alloc = [&](size_t bytes) -> void* {
        void* p = wsp + off;
        off = (off + bytes + 255) & ~(size_t)255;
        return p;
    };
    _Float16* embF16   = (_Float16*)alloc((size_t)V_ * E_ * 2);
    _Float16* Bp_wih0  = (_Float16*)alloc((size_t)G4_ * E_ * 2);
    float*    Pvoc     = (float*)   alloc((size_t)V_ * G4_ * 4);
    _Float16* Bp_hh0   = (_Float16*)alloc((size_t)G4_ * HT_ * 2);
    _Float16* Bp_cat1  = (_Float16*)alloc((size_t)G4_ * 1024 * 2);
    _Float16* Bp_hid   = (_Float16*)alloc((size_t)E_ * 1024 * 2);
    _Float16* Bp_emb   = (_Float16*)alloc((size_t)V_ * E_ * 2);
    _Float16* Bp_wa    = (_Float16*)alloc((size_t)HS_ * HT_ * 2);
    _Float16* srcoutH  = (_Float16*)alloc((size_t)S_ * B_ * HS_ * 2);
    _Float16* WhsH     = (_Float16*)alloc((size_t)S_ * B_ * HT_ * 2);
    _Float16* H0f16    = (_Float16*)alloc((size_t)B_ * HT_ * 2);
    _Float16* Xcat1    = (_Float16*)alloc((size_t)B_ * 1024 * 2);
    _Float16* XcatH    = (_Float16*)alloc((size_t)B_ * 1024 * 2);
    float*    G0       = (float*)   alloc((size_t)B_ * G4_ * 4);
    float*    G1       = (float*)   alloc((size_t)B_ * G4_ * 4);
    float*    cst      = (float*)   alloc((size_t)2 * B_ * HT_ * 4);
    float*    htf32    = (float*)   alloc((size_t)B_ * HT_ * 4);
    float*    scores   = (float*)   alloc((size_t)S_ * B_ * 4);
    float*    almt     = (float*)   alloc((size_t)S_ * B_ * 4);
    float*    ctx_emb  = (float*)   alloc((size_t)B_ * E_ * 4);
    float*    hidraw   = (float*)   alloc((size_t)B_ * E_ * 4);
    _Float16* hidresH  = (_Float16*)alloc((size_t)B_ * E_ * 2);
    float*    logits   = (float*)   alloc((size_t)B_ * V_ * 4);

    // ---- Precompute (every call; deterministic) ----
    f32_to_f16_kernel<<<(V_ * E_ + 255) / 256, 256, 0, stream>>>(emb, embF16, V_ * E_);
    pack_b_kernel<<<(G4_ / 16) * (E_ / 32) * 32 / 256, 256, 0, stream>>>(W_ih0, Bp_wih0, G4_, E_, E_ / 32, 0, 0);
    launch_gemm<float>(embF16, Bp_wih0, Pvoc, V_, G4_, E_, stream);   // P_vocab = emb @ W_ih0^T
    pack_b_kernel<<<(G4_ / 16) * (HT_ / 32) * 32 / 256, 256, 0, stream>>>(W_hh0, Bp_hh0, G4_, HT_, HT_ / 32, 0, 0);
    pack_b_kernel<<<(G4_ / 16) * (HT_ / 32) * 32 / 256, 256, 0, stream>>>(W_ih1, Bp_cat1, G4_, HT_, 32, 0, 0);
    pack_b_kernel<<<(G4_ / 16) * (HT_ / 32) * 32 / 256, 256, 0, stream>>>(W_hh1, Bp_cat1, G4_, HT_, 32, 16, 0);
    pack_b_kernel<<<(E_ / 16) * (1024 / 32) * 32 / 256, 256, 0, stream>>>(W_hidden, Bp_hid, E_, 1024, 1024 / 32, 0, 0);
    pack_b_kernel<<<(V_ / 16) * (E_ / 32) * 32 / 256, 256, 0, stream>>>(emb, Bp_emb, V_, E_, E_ / 32, 0, 0);
    pack_b_kernel<<<(HT_ / 16) * (HS_ / 32) * 32 / 256, 256, 0, stream>>>(Wa, Bp_wa, HT_, HS_, HS_ / 32, 0, 1);
    f32_to_f16_kernel<<<(S_ * B_ * HS_ + 255) / 256, 256, 0, stream>>>(src_out, srcoutH, S_ * B_ * HS_);
    // Hoisted Wh_s = src_outputs @ Wa : [S*B, HS] x [HS, HT]
    launch_gemm<_Float16>(srcoutH, Bp_wa, WhsH, S_ * B_, HT_, HS_, stream);
    init_state_kernel<<<(B_ * HT_ + 255) / 256, 256, 0, stream>>>(h0, c0, cst, H0f16, Xcat1);

    // ---- Decode loop (recurrent; stream-ordered) ----
    for (int t = 0; t < T_; ++t) {
        // layer 0 gates: G0 = h0 @ W_hh0^T   (emb part comes from Pvoc gather)
        launch_gemm<float>(H0f16, Bp_hh0, G0, B_, G4_, HT_, stream);
        cell0_kernel<<<(B_ * HT_ + 255) / 256, 256, 0, stream>>>(
            G0, Pvoc, target, sot, b_ih0, b_hh0, cst, H0f16, Xcat1, t);
        // layer 1 gates: G1 = [h0n | h1] @ [W_ih1 | W_hh1]^T
        launch_gemm<float>(Xcat1, Bp_cat1, G1, B_, G4_, 1024, stream);
        cell1_kernel<<<(B_ * HT_ + 255) / 256, 256, 0, stream>>>(
            G1, b_ih1, b_hh1, cst, Xcat1, XcatH, htf32);
        // attention
        scores_kernel<<<(S_ * B_ * 32) / 256, 256, 0, stream>>>(WhsH, htf32, mask, scores);
        softmax_kernel<<<B_, S_, 0, stream>>>(scores, almt);
        ctx_kernel<<<B_, 256, 0, stream>>>(almt, src_out, src_emb, XcatH, ctx_emb);
        // hidden projection: [ht | ctx] @ W_hidden^T
        launch_gemm<float>(XcatH, Bp_hid, hidraw, B_, E_, 1024, stream);
        residual_kernel<<<B_, E_, 0, stream>>>(hidraw, b_hidden, ctx_emb, hidresH);
        // logits: hid_res @ emb^T
        launch_gemm<float>(hidresH, Bp_emb, logits, B_, V_, E_, stream);
        logsoftmax_kernel<<<B_, V_, 0, stream>>>(logits, out, t);
    }
}
